// MemoryLSTM_7980049236087
// MI455X (gfx1250) — matrix operand order repaired and checked by reading, not compiled, not run on hardware
//
#include <hip/hip_runtime.h>
#include <hip/hip_bf16.h>

// ---------------------------------------------------------------------------
// MemoryLSTM for MI455X (gfx1250, wave32, WMMA + async global->LDS DMA)
//   B=16, T=2048, D=H=512, gates 4H=2048
// ---------------------------------------------------------------------------

typedef __attribute__((ext_vector_type(16))) _Float16 v16h;
typedef __attribute__((ext_vector_type(8)))  _Float16 v8h;
typedef __attribute__((ext_vector_type(8)))  float    v8f;

#define LSTM_B   16
#define LSTM_T   2048
#define LSTM_H   512
#define LSTM_G4  2048              // 4*H
#define LSTM_M   (LSTM_B * LSTM_T) // 32768 flattened rows
#define GX_BLK   (LSTM_G4 * LSTM_B)   // floats per timestep block (128KB)

// ---------------- helpers ----------------

__device__ __forceinline__ float sigmoid_(float x) {
    return 1.0f / (1.0f + __expf(-x));
}
__device__ __forceinline__ float tanh_(float x) {
    float ax = fabsf(x);
    float e  = __expf(-2.0f * ax);
    float t  = (1.0f - e) / (1.0f + e);
    return copysignf(t, x);
}

// A-fragment (16x32 f16): lane holds row (lane&15);
// halves 0..15 = K[kg..kg+15], kg=(lane<16?0:16).
// p must point at rowBase + k0 + kg.
__device__ __forceinline__ v16h make_a_frag(const _Float16* p) {
    v8h lo = *(const v8h*)(p);
    v8h hi = *(const v8h*)(p + 8);
    v16h a;
#pragma unroll
    for (int i = 0; i < 8; ++i) { a[i] = lo[i]; a[i + 8] = hi[i]; }
    return a;
}

// B-fragment (32x16 f16): lane holds column (lane&15), 16 contiguous K halves
// starting at K = (lane<16 ? 0 : 16). p must point at colRow + (hi?16:0).
__device__ __forceinline__ v16h ldb(const _Float16* p) {
    return *(const v16h*)p;       // 32B contiguous
}

#define WMMA(a, b, c) __builtin_amdgcn_wmma_f32_16x16x32_f16( \
        false, (a), false, (b), (short)0, (c), false, false)

// LDS byte offset of a __shared__ object = low 32 bits of its generic address
// (flat aperture rule, CDNA5 ISA 10.2).
__device__ __forceinline__ unsigned lds_off(const void* p) {
    return (unsigned)(unsigned long long)p;
}

// one 16B async global->LDS copy per lane (ASYNCcnt-tracked DMA)
__device__ __forceinline__ void async_g2l_b128(unsigned lds_addr, const void* g) {
    asm volatile("global_load_async_to_lds_b128 %0, %1, off"
                 :
                 : "v"(lds_addr), "v"((unsigned long long)g)
                 : "memory");
}
__device__ __forceinline__ void wait_asynccnt0() {
    asm volatile("s_wait_asynccnt 0x0" ::: "memory");
}

// ---------------- small prep kernels ----------------

__global__ void cvt_f16_kernel(const float* __restrict__ src,
                               _Float16* __restrict__ dst, int n) {
    int i = blockIdx.x * blockDim.x + threadIdx.x;
    int stride = gridDim.x * blockDim.x;
    for (; i < n; i += stride) dst[i] = (_Float16)src[i];
}

__global__ void bias_sum_kernel(const float* __restrict__ a,
                                const float* __restrict__ b,
                                float* __restrict__ o, int n) {
    int i = blockIdx.x * blockDim.x + threadIdx.x;
    if (i < n) o[i] = a[i] + b[i];
}

// ---------------- phase A: gx = A @ W^T + bias ----------------
// A   : [32768][512] f16, row-major (m = b*T + t)
// W   : [2048][512]  f16, row-major
// gx  : [T][4H][B]   f32  (batch innermost -> recurrence reads are v8f loads)
// Each wave computes a 16(M) x 64(N) tile, software-pipelined over k.
__global__ __launch_bounds__(256) void gemm_gx_kernel(
    const _Float16* __restrict__ A,
    const _Float16* __restrict__ W,
    const float* __restrict__ bias,
    float* __restrict__ gx)
{
    const int tid   = threadIdx.x;
    const int lane  = tid & 31;
    const int ln    = lane & 15;
    const int hi    = lane >> 4;            // 0 or 1
    const int gwave = (blockIdx.x * 256 + tid) >> 5;  // 0..65535
    const int mtile = gwave >> 5;           // 0..2047
    const int ngrp  = gwave & 31;           // 0..31
    const int n0    = ngrp * 64;

    const int row = mtile * 16 + ln;        // A row for this lane's fragment
    const _Float16* ap0 = A + (size_t)row * LSTM_H + hi * 16;
    const _Float16* wp0 = W + (size_t)(n0 +  0 + ln) * LSTM_H + hi * 16;
    const _Float16* wp1 = W + (size_t)(n0 + 16 + ln) * LSTM_H + hi * 16;
    const _Float16* wp2 = W + (size_t)(n0 + 32 + ln) * LSTM_H + hi * 16;
    const _Float16* wp3 = W + (size_t)(n0 + 48 + ln) * LSTM_H + hi * 16;

    v8f acc0 = {}, acc1 = {}, acc2 = {}, acc3 = {};

    // software pipeline: fetch k-step i+1 while issuing WMMAs of k-step i
    v16h a  = make_a_frag(ap0);
    v16h b0 = ldb(wp0), b1 = ldb(wp1), b2 = ldb(wp2), b3 = ldb(wp3);
#pragma unroll
    for (int k0 = 32; k0 <= LSTM_H; k0 += 32) {
        v16h an = a, n0f = b0, n1f = b1, n2f = b2, n3f = b3;
        if (k0 < LSTM_H) {
            an  = make_a_frag(ap0 + k0);
            n0f = ldb(wp0 + k0); n1f = ldb(wp1 + k0);
            n2f = ldb(wp2 + k0); n3f = ldb(wp3 + k0);
        }
        acc0 = WMMA(a, b0, acc0);
        acc1 = WMMA(a, b1, acc1);
        acc2 = WMMA(a, b2, acc2);
        acc3 = WMMA(a, b3, acc3);
        a = an; b0 = n0f; b1 = n1f; b2 = n2f; b3 = n3f;
    }

    // epilogue: +bias, scatter into [T][4H][B]
    v8f accv[4] = {acc0, acc1, acc2, acc3};
#pragma unroll
    for (int i = 0; i < 4; ++i) {
        const int coln = n0 + i * 16 + ln;
        const float bv = bias[coln];
#pragma unroll
        for (int r = 0; r < 8; ++r) {
            int m = mtile * 16 + r + hi * 8;   // flattened (b,t)
            int b = m >> 11;                   // / T
            int t = m & (LSTM_T - 1);
            gx[((size_t)t * LSTM_G4 + coln) * LSTM_B + b] = accv[i][r] + bv;
        }
    }
}

// ---------------- phase B: sequential LSTM recurrence ----------------
// Single persistent workgroup: 1024 threads = 32 waves on one WGP.
// Wave w owns h-columns [16w, 16w+16); per step it computes the 4 coupled
// gate tiles (i at n=col, f at 512+col, g at 1024+col, o at 1536+col).
// h ping-pongs in LDS (f16, A-matrix layout); c stays in VGPRs in the
// WMMA C/D layout (lane = column, elem r = batch r + 8*hi).
// gx[t] blocks (128KB each) are double-buffered in LDS via async
// global->LDS DMA issued one timestep ahead (ASYNCcnt), so the HBM
// latency of gx never sits on the serial path.
// LDS: 32KB h ping-pong + 2*128KB gx staging = 288KB (<320KB WGP limit).
__global__ __launch_bounds__(1024) void lstm_rec_kernel(
    const float*    __restrict__ gx,       // [T][4H][B] f32 (bias folded in)
    const _Float16* __restrict__ whh,      // [4H][H] f16
    _Float16*       __restrict__ inner_h,  // layer1: [B][T][H] f16 (else null)
    const float*    __restrict__ residual, // layer2: feats f32 (else null)
    float*          __restrict__ out,      // layer2: [B][T][H] f32 (else null)
    float*          __restrict__ hT,       // [B][H] f32
    float*          __restrict__ cT)       // [B][H] f32
{
    __shared__ _Float16 hbuf[2][LSTM_B][LSTM_H];   // 2 * 16KB ping-pong
    __shared__ float    gbuf[2][GX_BLK];           // 2 * 128KB gx staging

    const int tid  = threadIdx.x;
    const int wave = tid >> 5;
    const int lane = tid & 31;
    const int ln   = lane & 15;
    const int hi   = lane >> 4;
    const int col  = wave * 16 + ln;       // owned hidden column

    // h0 = 0
    for (int i = tid; i < LSTM_B * LSTM_H; i += 1024)
        ((_Float16*)hbuf[0])[i] = (_Float16)0.0f;

    // stage gx[0] into gbuf[0]: 1024 threads x 8 x 16B = 128KB
    {
        const char* src = (const char*)gx;
        unsigned    dst = lds_off(&gbuf[0][0]);
#pragma unroll
        for (int r = 0; r < 8; ++r)
            async_g2l_b128(dst + r * 16384 + tid * 16, src + r * 16384 + tid * 16);
    }

    v8f c  = {};   // cell state, elem r -> (batch = r + 8*hi, col)
    v8f h8 = {};   // last hidden value (f32)

    // Whh fragment base pointers for the 4 gates
    const _Float16* bp0 = whh + (size_t)(0 * LSTM_H + col) * LSTM_H + hi * 16;
    const _Float16* bp1 = whh + (size_t)(1 * LSTM_H + col) * LSTM_H + hi * 16;
    const _Float16* bp2 = whh + (size_t)(2 * LSTM_H + col) * LSTM_H + hi * 16;
    const _Float16* bp3 = whh + (size_t)(3 * LSTM_H + col) * LSTM_H + hi * 16;

    wait_asynccnt0();
    __syncthreads();

    int p = 0;
    for (int t = 0; t < LSTM_T; ++t) {
        const int cur = t & 1;
        // kick off async DMA of gx[t+1] into the other LDS buffer; it
        // overlaps with this step's WMMA work and is fenced by the
        // end-of-step s_wait_asynccnt + barrier.
        if (t + 1 < LSTM_T) {
            const char* src = (const char*)(gx + (size_t)(t + 1) * GX_BLK);
            unsigned    dst = lds_off(&gbuf[cur ^ 1][0]);
#pragma unroll
            for (int r = 0; r < 8; ++r)
                async_g2l_b128(dst + r * 16384 + tid * 16, src + r * 16384 + tid * 16);
        }

        // init accumulators from staged input gates: one v8f (LDS) per gate
        const float* grow = &gbuf[cur][0];
        v8f acc0 = *(const v8f*)(grow + ((size_t)(0 * LSTM_H) + col) * LSTM_B + hi * 8);
        v8f acc1 = *(const v8f*)(grow + ((size_t)(1 * LSTM_H) + col) * LSTM_B + hi * 8);
        v8f acc2 = *(const v8f*)(grow + ((size_t)(2 * LSTM_H) + col) * LSTM_B + hi * 8);
        v8f acc3 = *(const v8f*)(grow + ((size_t)(3 * LSTM_H) + col) * LSTM_B + hi * 8);

        // g += h @ Whh^T   (16 k-steps x 4 gates = 64 WMMA per wave per step)
#pragma unroll
        for (int k0 = 0; k0 < LSTM_H; k0 += 32) {
            v16h a  = make_a_frag(&hbuf[p][ln][k0 + hi * 16]);
            v16h b0 = ldb(bp0 + k0);
            v16h b1 = ldb(bp1 + k0);
            v16h b2 = ldb(bp2 + k0);
            v16h b3 = ldb(bp3 + k0);
            acc0 = WMMA(a, b0, acc0);
            acc1 = WMMA(a, b1, acc1);
            acc2 = WMMA(a, b2, acc2);
            acc3 = WMMA(a, b3, acc3);
        }

        // gate nonlinearities + state update (pure per-lane VALU)
#pragma unroll
        for (int r = 0; r < 8; ++r) {
            float iv = sigmoid_(acc0[r]);
            float fv = sigmoid_(acc1[r]);
            float gv = tanh_(acc2[r]);
            float ov = sigmoid_(acc3[r]);
            float cn = fv * c[r] + iv * gv;
            c[r]  = cn;
            h8[r] = ov * tanh_(cn);
        }

        // publish h: LDS ping-pong buffer (f16) + per-layer global output
#pragma unroll
        for (int r = 0; r < 8; ++r) {
            int b = r + hi * 8;
            _Float16 hh = (_Float16)h8[r];
            hbuf[p ^ 1][b][col] = hh;
            size_t idx = ((size_t)b * LSTM_T + t) * LSTM_H + col;
            if (inner_h) inner_h[idx] = hh;
            if (out)     out[idx] = h8[r] + residual[idx];
        }

        wait_asynccnt0();   // gx[t+1] fully landed in LDS (per-wave)
        __syncthreads();    // visible to all waves; h ping-pong complete
        p ^= 1;
    }

    // final hT / cT
#pragma unroll
    for (int r = 0; r < 8; ++r) {
        int b = r + hi * 8;
        hT[(size_t)b * LSTM_H + col] = h8[r];
        cT[(size_t)b * LSTM_H + col] = c[r];
    }
}

// ---------------- host-side orchestration ----------------

extern "C" void kernel_launch(void* const* d_in, const int* in_sizes, int n_in,
                              void* d_out, int out_size, void* d_ws, size_t ws_size,
                              hipStream_t stream) {
    (void)in_sizes; (void)n_in; (void)out_size; (void)ws_size;

    const float* feats = (const float*)d_in[0];   // [16,2048,512]
    const float* Wih1  = (const float*)d_in[1];   // [2048,512]
    const float* Whh1  = (const float*)d_in[2];   // [2048,512]
    const float* bih1  = (const float*)d_in[3];
    const float* bhh1  = (const float*)d_in[4];
    const float* Wih2  = (const float*)d_in[5];
    const float* Whh2  = (const float*)d_in[6];
    const float* bih2  = (const float*)d_in[7];
    const float* bhh2  = (const float*)d_in[8];

    float* outp = (float*)d_out;                   // out | h1 | c1 | h2 | c2
    float* out_seq = outp;                         // 16*2048*512
    float* h1 = outp + (size_t)LSTM_B * LSTM_T * LSTM_H;
    float* c1 = h1 + LSTM_B * LSTM_H;
    float* h2 = c1 + LSTM_B * LSTM_H;
    float* c2 = h2 + LSTM_B * LSTM_H;

    // workspace layout
    char* ws = (char*)d_ws;
    size_t off = 0;
    auto alloc = [&](size_t bytes) {
        void* p = ws + off;
        off = (off + bytes + 255) & ~(size_t)255;
        return p;
    };
    _Float16* feats_h = (_Float16*)alloc((size_t)LSTM_M * LSTM_H * 2);   // 32 MB
    _Float16* wih1_h  = (_Float16*)alloc((size_t)LSTM_G4 * LSTM_H * 2);  // 2 MB
    _Float16* whh1_h  = (_Float16*)alloc((size_t)LSTM_G4 * LSTM_H * 2);
    _Float16* wih2_h  = (_Float16*)alloc((size_t)LSTM_G4 * LSTM_H * 2);
    _Float16* whh2_h  = (_Float16*)alloc((size_t)LSTM_G4 * LSTM_H * 2);
    float*    b1      = (float*)alloc(LSTM_G4 * 4);
    float*    b2      = (float*)alloc(LSTM_G4 * 4);
    _Float16* inner_h = (_Float16*)alloc((size_t)LSTM_M * LSTM_H * 2);   // 32 MB
    float*    gx      = (float*)alloc((size_t)LSTM_T * GX_BLK * 4);      // 256 MB

    const int NW = LSTM_G4 * LSTM_H;          // weight element count
    const int NF = LSTM_M * LSTM_H;           // feats element count

    // f32 -> f16 conversions
    cvt_f16_kernel<<<2048, 256, 0, stream>>>(feats, feats_h, NF);
    cvt_f16_kernel<<<1024, 256, 0, stream>>>(Wih1, wih1_h, NW);
    cvt_f16_kernel<<<1024, 256, 0, stream>>>(Whh1, whh1_h, NW);
    cvt_f16_kernel<<<1024, 256, 0, stream>>>(Wih2, wih2_h, NW);
    cvt_f16_kernel<<<1024, 256, 0, stream>>>(Whh2, whh2_h, NW);
    // fused biases
    bias_sum_kernel<<<8, 256, 0, stream>>>(bih1, bhh1, b1, LSTM_G4);
    bias_sum_kernel<<<8, 256, 0, stream>>>(bih2, bhh2, b2, LSTM_G4);

    // 65536 waves total -> 8192 blocks of 8 waves
    const int gemm_blocks = (LSTM_M / 16) * ((LSTM_G4 / 16) / 4) / 8;

    // ---- layer 1 ----
    gemm_gx_kernel<<<gemm_blocks, 256, 0, stream>>>(feats_h, wih1_h, b1, gx);
    lstm_rec_kernel<<<1, 1024, 0, stream>>>(gx, whh1_h, inner_h,
                                            nullptr, nullptr, h1, c1);
    // ---- layer 2 ----
    gemm_gx_kernel<<<gemm_blocks, 256, 0, stream>>>(inner_h, wih2_h, b2, gx);
    lstm_rec_kernel<<<1, 1024, 0, stream>>>(gx, whh2_h, nullptr,
                                            feats, out_seq, h2, c2);
}
